// LowRankRNN_91036126806947
// MI455X (gfx1250) — matrix operand order repaired and checked by reading, not compiled, not run on hardware
//
#include <hip/hip_runtime.h>
#include <hip/hip_bf16.h>

typedef __attribute__((ext_vector_type(16))) _Float16 v16h;
typedef __attribute__((ext_vector_type(8)))  _Float16 v8h;
typedef __attribute__((ext_vector_type(8)))  float    v8f;

namespace {
constexpr int Hdim = 2048;
constexpr int Bdim = 256;
constexpr int Tdim = 300;
constexpr int Idim = 4;
constexpr int Odim = 2;
constexpr float kNoiseStd = 0.05f;
constexpr float kAlpha    = 0.2f;
}

// ---------------------------------------------------------------- zero d_out
__global__ void lrr_zero_f32(float* __restrict__ p, int n) {
  int i = blockIdx.x * blockDim.x + threadIdx.x;
  if (i < n) p[i] = 0.0f;
}

// -------------------------------------------- Wt[n][k] = wrec[n][k] in fp16
// wrec[i][j] = m[i,0]*n[j,0] + m[i,1]*n[j,1] + rec_noise[i][j]
// Stored row-major with K contiguous -> WMMA B-fragment loads are contiguous.
__global__ __launch_bounds__(256) void lrr_prep_w(
    const float* __restrict__ m2, const float* __restrict__ n2,
    const float* __restrict__ rec, _Float16* __restrict__ Wt) {
  int idx = blockIdx.x * 256 + threadIdx.x;     // idx = i*H + j
  int i = idx >> 11;
  int j = idx & (Hdim - 1);
  float v = m2[i * 2 + 0] * n2[j * 2 + 0] +
            m2[i * 2 + 1] * n2[j * 2 + 1] + rec[idx];
  Wt[idx] = (_Float16)v;
}

// -------------------------------- wi_full = wi*si[:,None], wo_full = wo*so
__global__ __launch_bounds__(256) void lrr_prep_small(
    const float* __restrict__ wi, const float* __restrict__ si,
    const float* __restrict__ wo, const float* __restrict__ so,
    float* __restrict__ wif, float* __restrict__ wof) {
  int idx = blockIdx.x * 256 + threadIdx.x;     // 0 .. 4*H + H*2 - 1
  if (idx < Idim * Hdim) {
    int c = idx / Hdim;
    wif[idx] = wi[idx] * si[c];
  } else {
    int k = idx - Idim * Hdim;                  // 0 .. H*O-1
    int o = k & 1;
    wof[k] = wo[k] * so[o];
  }
}

// ------------------------------------------------- h = h0 (bcast), r = tanh
__global__ __launch_bounds__(256) void lrr_init_state(
    const float* __restrict__ h0, float* __restrict__ h,
    _Float16* __restrict__ r0) {
  int idx = blockIdx.x * 256 + threadIdx.x;     // B*H
  int i = idx & (Hdim - 1);
  float v = h0[i];
  h[idx] = v;
  r0[idx] = (_Float16)tanhf(v);
}

// ----------------------------------------------------------- one scan step
// Grid: x = H/128 = 16 N-tiles, y = B/32 = 8 M-tiles. Block: 128 thr = 4 waves
// arranged 2(M) x 2(N); each wave computes a 16x64 strip = 4 WMMA accums.
__global__ __launch_bounds__(128) void lrr_step(
    const _Float16* __restrict__ Wt,    // [H][H] fp16 (N-major rows, K contig)
    const _Float16* __restrict__ rin,   // [B][H] fp16  (read)
    _Float16* __restrict__ rout,        // [B][H] fp16  (write)
    float* __restrict__ h,              // [B][H]
    const float* __restrict__ noise,    // [B][T][H]
    const float* __restrict__ inp,      // [B][T][I]
    const float* __restrict__ wif,      // [I][H]
    const float* __restrict__ wof,      // [H][O]
    float* __restrict__ out,            // [B][T][O]
    int t) {
  const int tid  = threadIdx.x;
  const int lane = tid & 31;
  const int wave = tid >> 5;            // 0..3
  const int mw = wave >> 1;             // 0..1
  const int nw = wave & 1;              // 0..1
  const int m0 = blockIdx.y * 32 + mw * 16;
  const int n0 = blockIdx.x * 128 + nw * 64;

  const int lh = lane & 15;             // position within half-wave
  const int hi = lane >> 4;             // 0 / 1

  // A-fragment (16-bit A 16x32): lane row = m0+lh;
  // elements 0..15 = K k0 + hi*16 + {0..15}.
  const _Float16* arow = rin + (size_t)(m0 + lh) * Hdim + hi * 16;

  v8f acc[4] = {v8f{0.f,0.f,0.f,0.f,0.f,0.f,0.f,0.f},
                v8f{0.f,0.f,0.f,0.f,0.f,0.f,0.f,0.f},
                v8f{0.f,0.f,0.f,0.f,0.f,0.f,0.f,0.f},
                v8f{0.f,0.f,0.f,0.f,0.f,0.f,0.f,0.f}};

  for (int k0 = 0; k0 < Hdim; k0 += 32) {
    __builtin_prefetch(arow + k0 + 32, 0, 3);   // WGP-scope global_prefetch_b8
    v8h alo = *(const v8h*)(arow + k0);
    v8h ahi = *(const v8h*)(arow + k0 + 8);
    v16h a = __builtin_shufflevector(alo, ahi, 0, 1, 2, 3, 4, 5, 6, 7,
                                     8, 9, 10, 11, 12, 13, 14, 15);
#pragma unroll
    for (int tn = 0; tn < 4; ++tn) {
      // B-fragment (16-bit B 32x16): lane col = n0+tn*16+lh,
      // 16 contiguous K halves starting at k0 + hi*16 (32B aligned).
      const _Float16* bp =
          Wt + (size_t)(n0 + tn * 16 + lh) * Hdim + k0 + hi * 16;
      v16h b = *(const v16h*)bp;
      acc[tn] = __builtin_amdgcn_wmma_f32_16x16x32_f16(
          false, a, false, b, (short)0, acc[tn], false, false);
    }
  }

  // Epilogue: h update, r=tanh(h), output projection (reduce over N).
#pragma unroll
  for (int v = 0; v < 8; ++v) {
    const int mrow = m0 + v + hi * 8;   // batch index b
    const float* xr = inp + ((size_t)mrow * Tdim + t) * Idim;
    const float x0 = __builtin_nontemporal_load(xr + 0);
    const float x1 = __builtin_nontemporal_load(xr + 1);
    const float x2 = __builtin_nontemporal_load(xr + 2);
    const float x3 = __builtin_nontemporal_load(xr + 3);
    float s0 = 0.f, s1 = 0.f;
#pragma unroll
    for (int tn = 0; tn < 4; ++tn) {
      const int n = n0 + tn * 16 + lh;
      const float accv = acc[tn][v];
      const float xw = x0 * wif[n] + x1 * wif[Hdim + n] +
                       x2 * wif[2 * Hdim + n] + x3 * wif[3 * Hdim + n];
      const size_t hidx = (size_t)mrow * Hdim + n;
      const float hold = h[hidx];
      // noise is a 629 MB stream read exactly once -> NT load, keep Wt/h in L2
      const float nz = __builtin_nontemporal_load(
          noise + ((size_t)mrow * Tdim + t) * Hdim + n);
      const float hnew =
          hold + kNoiseStd * nz + kAlpha * (-hold + accv + xw);
      h[hidx] = hnew;
      const float th = tanhf(hnew);
      rout[hidx] = (_Float16)th;
      s0 += th * wof[n * 2 + 0];
      s1 += th * wof[n * 2 + 1];
    }
    // reduce over the 16 lanes of this half-wave (same mrow)
#pragma unroll
    for (int off = 8; off >= 1; off >>= 1) {
      s0 += __shfl_xor(s0, off, 32);
      s1 += __shfl_xor(s1, off, 32);
    }
    if (lh == 0) {
      atomicAdd(&out[((size_t)mrow * Tdim + t) * Odim + 0], s0);
      atomicAdd(&out[((size_t)mrow * Tdim + t) * Odim + 1], s1);
    }
  }
}

// ---------------------------------------------------------------- launcher
extern "C" void kernel_launch(void* const* d_in, const int* in_sizes, int n_in,
                              void* d_out, int out_size, void* d_ws,
                              size_t ws_size, hipStream_t stream) {
  const float* inp   = (const float*)d_in[0];  // [B,T,I]
  const float* noise = (const float*)d_in[1];  // [B,T,H]
  const float* wi    = (const float*)d_in[2];  // [I,H]
  const float* si    = (const float*)d_in[3];  // [I]
  const float* m2    = (const float*)d_in[4];  // [H,R]
  const float* n2    = (const float*)d_in[5];  // [H,R]
  const float* rec   = (const float*)d_in[6];  // [H,H]
  const float* wo    = (const float*)d_in[7];  // [H,O]
  const float* so    = (const float*)d_in[8];  // [O]
  const float* h0    = (const float*)d_in[9];  // [H]
  float* out = (float*)d_out;

  char* ws = (char*)d_ws;
  _Float16* Wt  = (_Float16*)(ws);                       // 8 MB
  _Float16* rA  = (_Float16*)(ws + (8u << 20));          // 1 MB
  _Float16* rB  = (_Float16*)(ws + (9u << 20));          // 1 MB
  float*    h   = (float*)(ws + (10u << 20));            // 2 MB
  float*    wif = (float*)(ws + (12u << 20));            // 32 KB
  float*    wof = (float*)(ws + (12u << 20) + 65536);    // 16 KB

  lrr_zero_f32<<<(out_size + 255) / 256, 256, 0, stream>>>(out, out_size);
  lrr_prep_w<<<(Hdim * Hdim) / 256, 256, 0, stream>>>(m2, n2, rec, Wt);
  lrr_prep_small<<<(Idim * Hdim + Hdim * Odim) / 256, 256, 0, stream>>>(
      wi, si, wo, so, wif, wof);
  lrr_init_state<<<(Bdim * Hdim) / 256, 256, 0, stream>>>(h0, h, rA);

  for (int t = 0; t < Tdim; ++t) {
    const _Float16* rin = (t & 1) ? rB : rA;
    _Float16* rout      = (t & 1) ? rA : rB;
    lrr_step<<<dim3(Hdim / 128, Bdim / 32), 128, 0, stream>>>(
        Wt, rin, rout, h, noise, inp, wif, wof, out, t);
  }
}
